// CN_MLP_71631464563230
// MI455X (gfx1250) — matrix operand order repaired and checked by reading, not compiled, not run on hardware
//
#include <hip/hip_runtime.h>
#include <hip/hip_bf16.h>

// ---------------------------------------------------------------------------
// Problem sizes (compile-time constants from the reference)
// ---------------------------------------------------------------------------
#define BATCH  8192
#define D_IN   5000
#define KPAD1  5024      // D_IN padded to multiple of 32
#define HDIM   512
#define TTASK  16
#define KEXT   (TTASK * HDIM)   // 8192, K of the fused mixture GEMM

typedef __attribute__((ext_vector_type(16))) __bf16 v16bf;
typedef __attribute__((ext_vector_type(8)))  float  v8f;

static __device__ __forceinline__ __bf16 f2bf(float x) { return (__bf16)x; }

// ---------------------------------------------------------------------------
// CDNA5 async global->LDS DMA: copy 64 contiguous bytes (one tile row slice)
// per lane into LDS. INST_OFFSET is added to BOTH the LDS and global address
// (ISA 10. async), so one
// address pair serves all four 16B transfers. Tracked with ASYNCcnt.
//
// The LDS destination operand is derived by ptrtoint from a real pointer into
// the __shared__ buffer: flat shared addresses are SHARED_BASE|offset, so the
// low 32 bits are the LDS byte offset -- and the ptrtoint captures the LDS
// global, so the "memory" clobber forces the compiler to treat the buffer as
// written (otherwise GlobalOpt folds all reads of it to undef).
// ---------------------------------------------------------------------------
static __device__ __forceinline__ void async_copy_row64(void* ldsPtr,
                                                        const __bf16* g) {
    unsigned ldsOff = (unsigned)(unsigned long long)ldsPtr;  // LDS byte offset
    unsigned long long ga = (unsigned long long)g;
    asm volatile("global_load_async_to_lds_b128 %0, %1, off"
                 :: "v"(ldsOff), "v"(ga) : "memory");
    asm volatile("global_load_async_to_lds_b128 %0, %1, off offset:16"
                 :: "v"(ldsOff), "v"(ga) : "memory");
    asm volatile("global_load_async_to_lds_b128 %0, %1, off offset:32"
                 :: "v"(ldsOff), "v"(ga) : "memory");
    asm volatile("global_load_async_to_lds_b128 %0, %1, off offset:48"
                 :: "v"(ldsOff), "v"(ga) : "memory");
}

static __device__ __forceinline__ void wait_async0() {
    asm volatile("s_wait_asynccnt 0x0" ::: "memory");
}

// ---------------------------------------------------------------------------
// Convert fp32 -> bf16 with column zero-padding (row-major [rows, colsPad])
// ---------------------------------------------------------------------------
__global__ __launch_bounds__(256) void cvt_pad_bf16(const float* __restrict__ src,
                                                    __bf16* __restrict__ dst,
                                                    int cols, int colsPad) {
    int col = blockIdx.x * 256 + threadIdx.x;
    int row = blockIdx.y;
    if (col >= colsPad) return;
    float v = (col < cols) ? src[(size_t)row * cols + col] : 0.0f;
    dst[(size_t)row * colsPad + col] = f2bf(v);
}

// ---------------------------------------------------------------------------
// cmT[n][t*512+h] = CM[t][h][n]   ->  bf16 [512, 8192]
// ---------------------------------------------------------------------------
__global__ __launch_bounds__(256) void cvt_cm_transpose(const float* __restrict__ CM,
                                                        __bf16* __restrict__ cmT) {
    int kk = blockIdx.x * 256 + threadIdx.x;   // 0..8191 = t*512+h
    int n  = blockIdx.y;                       // 0..511
    int t = kk >> 9;
    int h = kk & 511;
    cmT[(size_t)n * KEXT + kk] = f2bf(CM[((size_t)t * HDIM + h) * HDIM + n]);
}

// ---------------------------------------------------------------------------
// WMMA bf16 GEMM:  C[M,N] = A[M,K] * Bt[N,K]^T  (+ bias), optional bf16 copy.
//
// Block = 256 threads = 8 waves; block tile 128(M) x 128(N); wave tile 64x32
// (4x2 of v_wmma_f32_16x16x32_bf16), waves arranged 2(m) x 4(n).
//
// K loop is double-buffered through LDS: each 32-K stage stages A[128x32] and
// B[128x32] bf16 (16 KB) via global_load_async_to_lds_b128 (ASYNCcnt), then
// waves read fragments with ds_load_b128. Pipeline per stage:
//      s_wait_asynccnt 0  ->  barrier  ->  issue stage i+1  ->  consume i
// Issuing after the barrier guarantees the buffer being overwritten has been
// fully consumed (its reads happened before the previous barrier).
//
// Fragment layouts per CDNA5 ISA 7.12.2 (16-bit A 16x32, B 32x16, f32 C/D).
// ---------------------------------------------------------------------------
#define STAGE_BYTES 16384            // 8 KB A + 8 KB B per stage
#define B_OFF       8192

__global__ __launch_bounds__(256) void gemm_bf16_wmma(
        const __bf16* __restrict__ A,    // [M, K] row-major
        const __bf16* __restrict__ Bt,   // [N, K] row-major (B transposed)
        float*        __restrict__ C,    // [M, N]
        __bf16*       __restrict__ Cbf,  // optional bf16 copy of C (or null)
        const float*  __restrict__ bias, // optional [N] (or null)
        int M, int N, int K) {
    __shared__ __align__(16) char smem[2 * STAGE_BYTES];

    const int tid  = threadIdx.x;
    const int lane = tid & 31;
    const int wave = tid >> 5;
    const int wm   = wave & 1;       // 0..1  (64 rows each)
    const int wn   = wave >> 1;      // 0..3  (32 cols each)
    const int mblk = blockIdx.x * 128;
    const int nblk = blockIdx.y * 128;

    const int l16 = lane & 15;
    const int hf  = lane >> 4;

    // --- staging assignment: threads 0..127 -> A rows, 128..255 -> B rows ---
    const bool isA = tid < 128;
    const int  srow = isA ? tid : (tid - 128);
    const __bf16* gsrc = isA ? (A  + (size_t)(mblk + srow) * K)
                             : (Bt + (size_t)(nblk + srow) * K);
    char* stageDst0 = smem + (isA ? 0 : B_OFF) + srow * 64;
    char* stageDst1 = stageDst0 + STAGE_BYTES;

    v8f acc[4][2] = {};

    const int nk = K / 32;
    // prologue: stage 0 into buffer 0
    async_copy_row64(stageDst0, gsrc);

    for (int i = 0; i < nk; ++i) {
        wait_async0();         // my stage-i async transfers are complete
        __syncthreads();       // everyone's are; previous buffer fully drained

        if (i + 1 < nk) {
            async_copy_row64(((i + 1) & 1) ? stageDst1 : stageDst0,
                             gsrc + (i + 1) * 32);
            // pull the stage-after-next toward L2 (global_prefetch_b8)
            __builtin_prefetch(gsrc + (i + 3) * 32, 0, 1);
        }

        const char* buf = smem + (i & 1) * STAGE_BYTES;
        const __bf16* ldsA = (const __bf16*)buf;            // [128][32]
        const __bf16* ldsB = (const __bf16*)(buf + B_OFF);  // [128][32]

        union { v16bf v; uint4 q[2]; } af[4], bfr[2];
        #pragma unroll
        for (int mi = 0; mi < 4; ++mi) {
            const __bf16* r = ldsA + (wm * 64 + mi * 16 + l16) * 32;
            af[mi].q[0] = *(const uint4*)(r +      hf * 8);
            af[mi].q[1] = *(const uint4*)(r + 16 + hf * 8);
        }
        #pragma unroll
        for (int ni = 0; ni < 2; ++ni) {
            const __bf16* r = ldsB + (wn * 32 + ni * 16 + l16) * 32;
            bfr[ni].q[0] = *(const uint4*)(r +      hf * 8);
            bfr[ni].q[1] = *(const uint4*)(r + 16 + hf * 8);
        }
        #pragma unroll
        for (int mi = 0; mi < 4; ++mi)
            #pragma unroll
            for (int ni = 0; ni < 2; ++ni)
                acc[mi][ni] = __builtin_amdgcn_wmma_f32_16x16x32_bf16(
                        false, af[mi].v, false, bfr[ni].v,
                        (short)0, acc[mi][ni], false, false);
    }

    // Epilogue: C/D layout -> lane holds col n = l16, rows v + 8*hf per VGPR v
    #pragma unroll
    for (int mi = 0; mi < 4; ++mi) {
        #pragma unroll
        for (int ni = 0; ni < 2; ++ni) {
            const int n  = nblk + wn * 32 + ni * 16 + l16;
            const float bv = bias ? bias[n] : 0.0f;
            #pragma unroll
            for (int v = 0; v < 8; ++v) {
                const int m  = mblk + wm * 64 + mi * 16 + hf * 8 + v;
                const float val = acc[mi][ni][v] + bv;
                C[(size_t)m * N + n] = val;
                if (Cbf) Cbf[(size_t)m * N + n] = f2bf(val);
            }
        }
    }
}

// ---------------------------------------------------------------------------
// Task-attention softmax: w[b,t] = softmax_t( mlp[b,:] @ attn[:,t] ), T=16
// One wave32 per row; lane t<16 owns task t (lanes 16..31 mirror).
// ---------------------------------------------------------------------------
__global__ __launch_bounds__(256) void attn_softmax(const float* __restrict__ mlp,
                                                    const float* __restrict__ attn,
                                                    float* __restrict__ wsoft) {
    const int b    = blockIdx.x * 8 + (threadIdx.x >> 5);
    const int lane = threadIdx.x & 31;
    const int t    = lane & 15;
    const float* row = mlp + (size_t)b * HDIM;
    float d = 0.0f;
    for (int h = 0; h < HDIM; ++h)
        d = fmaf(row[h], attn[h * TTASK + t], d);
    float mx = d;
    for (int off = 8; off; off >>= 1) mx = fmaxf(mx, __shfl_xor(mx, off, 16));
    const float e = __expf(d - mx);
    float s = e;
    for (int off = 8; off; off >>= 1) s += __shfl_xor(s, off, 16);
    if (lane < 16) wsoft[(size_t)b * TTASK + t] = e / s;
}

// ---------------------------------------------------------------------------
// Aext[b, t*512+h] = w[b,t] * mlp[b,h]   (bf16 [8192, 8192])
// ---------------------------------------------------------------------------
__global__ __launch_bounds__(256) void build_aext(const float* __restrict__ mlp,
                                                  const float* __restrict__ wsoft,
                                                  __bf16* __restrict__ aext) {
    const long idx = (long)blockIdx.x * 256 + threadIdx.x;  // b*8192 + kk
    const int b  = (int)(idx >> 13);
    const int kk = (int)(idx & 8191);
    const int t  = kk >> 9;
    const int h  = kk & 511;
    aext[idx] = f2bf(wsoft[(size_t)b * TTASK + t] * mlp[(size_t)b * HDIM + h]);
}

// ---------------------------------------------------------------------------
// Classifier: out[b] = sigmoid( out1[b,:] @ cla_w + cla_b ), wave per row.
// ---------------------------------------------------------------------------
__global__ __launch_bounds__(256) void classifier(const float* __restrict__ out1,
                                                  const float* __restrict__ cw,
                                                  const float* __restrict__ cb,
                                                  float* __restrict__ out) {
    const int b    = blockIdx.x * 8 + (threadIdx.x >> 5);
    const int lane = threadIdx.x & 31;
    const float* row = out1 + (size_t)b * HDIM;
    float s = 0.0f;
    for (int k = lane; k < HDIM; k += 32)
        s = fmaf(row[k], cw[k], s);
    for (int off = 16; off; off >>= 1) s += __shfl_xor(s, off, 32);
    if (lane == 0) out[b] = 1.0f / (1.0f + __expf(-(s + cb[0])));
}

// ---------------------------------------------------------------------------
// Launch
// ---------------------------------------------------------------------------
static size_t align256(size_t x) { return (x + 255) & ~(size_t)255; }

extern "C" void kernel_launch(void* const* d_in, const int* in_sizes, int n_in,
                              void* d_out, int out_size, void* d_ws, size_t ws_size,
                              hipStream_t stream) {
    const float* data  = (const float*)d_in[0];   // [8192, 5000]
    const float* mlp_w = (const float*)d_in[1];   // [512, 5000]
    const float* mlp_b = (const float*)d_in[2];   // [512]
    const float* CM    = (const float*)d_in[3];   // [16, 512, 512]
    const float* attn  = (const float*)d_in[4];   // [512, 16]
    const float* cla_w = (const float*)d_in[5];   // [1, 512]
    const float* cla_b = (const float*)d_in[6];   // [1]
    float*       out   = (float*)d_out;           // [8192]

    // Workspace carve-up (all 256B aligned)
    char* p = (char*)d_ws;
    size_t off = 0;
    __bf16* a_bf16  = (__bf16*)(p + off); off = align256(off + (size_t)BATCH * KPAD1 * 2);
    __bf16* w_bf16  = (__bf16*)(p + off); off = align256(off + (size_t)HDIM  * KPAD1 * 2);
    __bf16* cmT     = (__bf16*)(p + off); off = align256(off + (size_t)HDIM  * KEXT  * 2);
    __bf16* aext    = (__bf16*)(p + off); off = align256(off + (size_t)BATCH * KEXT  * 2);
    float*  mlp_f32 = (float*) (p + off); off = align256(off + (size_t)BATCH * HDIM  * 4);
    __bf16* mlp_bf  = (__bf16*)(p + off); off = align256(off + (size_t)BATCH * HDIM  * 2);
    float*  wsoft   = (float*) (p + off); off = align256(off + (size_t)BATCH * TTASK * 4);
    float*  out1    = (float*) (p + off); off = align256(off + (size_t)BATCH * HDIM  * 4);
    (void)ws_size; (void)in_sizes; (void)n_in; (void)out_size;

    // 1) Convert / pad inputs to bf16
    {
        dim3 g((KPAD1 + 255) / 256, BATCH);
        cvt_pad_bf16<<<g, 256, 0, stream>>>(data, a_bf16, D_IN, KPAD1);
    }
    {
        dim3 g((KPAD1 + 255) / 256, HDIM);
        cvt_pad_bf16<<<g, 256, 0, stream>>>(mlp_w, w_bf16, D_IN, KPAD1);
    }
    {
        dim3 g(KEXT / 256, HDIM);
        cvt_cm_transpose<<<g, 256, 0, stream>>>(CM, cmT);
    }

    // 2) MLP GEMM: mlp_out = data @ mlp_w^T + mlp_b   (f32 + bf16 copies)
    {
        dim3 g(BATCH / 128, HDIM / 128);
        gemm_bf16_wmma<<<g, 256, 0, stream>>>(a_bf16, w_bf16, mlp_f32, mlp_bf,
                                              mlp_b, BATCH, HDIM, KPAD1);
    }

    // 3) Softmax over tasks
    attn_softmax<<<BATCH / 8, 256, 0, stream>>>(mlp_f32, attn, wsoft);

    // 4) Aext[b,(t,h)] = w[b,t]*mlp[b,h]
    build_aext<<<(unsigned)((size_t)BATCH * KEXT / 256), 256, 0, stream>>>(mlp_f32, wsoft, aext);

    // 5) Fused mixture GEMM: out1 = Aext @ cmT^T   ([8192 x 8192 x 512])
    {
        dim3 g(BATCH / 128, HDIM / 128);
        gemm_bf16_wmma<<<g, 256, 0, stream>>>(aext, cmT, out1, ((__bf16*)nullptr),
                                              (const float*)nullptr, BATCH, HDIM, KEXT);
    }

    // 6) Classifier + sigmoid
    classifier<<<BATCH / 8, 256, 0, stream>>>(out1, cla_w, cla_b, out);
}
